// ScaledDotProductAttention_70987219468812
// MI455X (gfx1250) — matrix operand order repaired and checked by reading, not compiled, not run on hardware
//
#include <hip/hip_runtime.h>
#include <hip/hip_bf16.h>
#include <math.h>

// ---------------------------------------------------------------------------
// Fused scaled-dot-product attention for MI455X (gfx1250, wave32, WMMA).
//   B=16, S=2048, D=128, fp32 inputs.  Outputs: context [B,S,D] then attn
//   [B,S,S], concatenated flat in d_out.
//
// One workgroup (256 threads = 8 wave32) handles one (batch, 16-query-row)
// tile.  Math path: fp32 -> f16 WMMA (v_wmma_f32_16x16x32_f16) with fp32
// accumulation; softmax in fp32.  Scores live in LDS [16][2052] fp32.
// All wave-dependent control flow is forced scalar via readfirstlane so the
// WMMA regions compile with uniform (EXEC = all-ones) flow.
// ---------------------------------------------------------------------------

#define BATCH      16
#define SEQ        2048
#define DIM        128
#define QTILE      16
#define WAVES      8
#define THREADS    256
#define KTILES     (SEQ / 16)          // 128 key tiles of 16
#define ROWSTRIDE  2052                // 2048 + 4 pad (4 mod 64 -> conflict-free)
#define LDS_BYTES  (QTILE * ROWSTRIDE * 4)

typedef __attribute__((ext_vector_type(16))) _Float16 v16h;
typedef __attribute__((ext_vector_type(8)))  float    v8f;

__device__ __forceinline__ v16h pack_h16(float4 a, float4 b, float4 c, float4 d) {
    v16h t;
    t[0]  = (_Float16)a.x; t[1]  = (_Float16)a.y; t[2]  = (_Float16)a.z; t[3]  = (_Float16)a.w;
    t[4]  = (_Float16)b.x; t[5]  = (_Float16)b.y; t[6]  = (_Float16)b.z; t[7]  = (_Float16)b.w;
    t[8]  = (_Float16)c.x; t[9]  = (_Float16)c.y; t[10] = (_Float16)c.z; t[11] = (_Float16)c.w;
    t[12] = (_Float16)d.x; t[13] = (_Float16)d.y; t[14] = (_Float16)d.z; t[15] = (_Float16)d.w;
    return t;
}

__global__ void __launch_bounds__(THREADS)
sdpa_wmma_kernel(const float* __restrict__ Q, const float* __restrict__ K,
                 const float* __restrict__ V, float* __restrict__ ctx,
                 float* __restrict__ attn) {
    extern __shared__ float lds[];               // scores/probs [QTILE][ROWSTRIDE]

    const int b      = blockIdx.x >> 7;          // / (SEQ/QTILE)
    const int qt     = blockIdx.x & 127;
    const int tid    = threadIdx.x;
    const int lane   = tid & 31;
    const int lane_m = lane & 15;                // row (A/C) or column (B)
    const int lane_h = lane >> 4;                // lane half
    // wave index as a true SGPR -> all wave-level loops/branches are scalar
    const int wave   = __builtin_amdgcn_readfirstlane(tid >> 5);

    const float* Qb = Q + ((size_t)b * SEQ + (size_t)qt * QTILE) * DIM;
    const float* Kb = K + (size_t)b * SEQ * DIM;
    const float* Vb = V + (size_t)b * SEQ * DIM;

    // ---- Q tile -> four 16x32 f16 A-fragments (held in registers) ----------
    // ISA 16-bit A layout: lane holds row m=lane%16; element e maps to
    // K = 32*d + 16*(e>>3) + 8*(lane>>4) + (e&7).
    v16h aq[4];
    {
        const float* qrow = Qb + lane_m * DIM + 8 * lane_h;
#pragma unroll
        for (int d = 0; d < 4; ++d) {
            const float4* q0 = reinterpret_cast<const float4*>(qrow + 32 * d);
            const float4* q1 = reinterpret_cast<const float4*>(qrow + 32 * d + 16);
            aq[d] = pack_h16(q0[0], q0[1], q1[0], q1[1]);
        }
    }

    // ---- Pass 1: S = scale * Q K^T, scores -> LDS --------------------------
    // ISA 16-bit B layout (32x16): lane holds column n=lane%16;
    // element e maps to K = 32*d + 16*(e>>3) + 8*(lane>>4) + (e&7).
    const float scale = 0.088388347648318447f;   // 1/sqrt(128)
    for (int t = wave; t < KTILES; t += WAVES) { // scalar loop (t uniform)
        const int key0 = t * 16;
        const float* krow = Kb + (size_t)(key0 + lane_m) * DIM + 8 * lane_h;
        if (t + WAVES < KTILES) {                // scalar-guarded prefetch
            __builtin_prefetch(Kb + (size_t)(key0 + WAVES * 16 + lane_m) * DIM
                                   + 16 * lane_h, 0, 1);
        }
        v8f c = {};
#pragma unroll
        for (int d = 0; d < 4; ++d) {
            const float4* kp = reinterpret_cast<const float4*>(krow + 32 * d);
            v16h bk = pack_h16(kp[0], kp[1], kp[4], kp[5]);
            c = __builtin_amdgcn_wmma_f32_16x16x32_f16(
                    false, aq[d], false, bk, (short)0, c, false, false);
        }
        // C layout: VGPR r -> row m = r + 8*(lane>>4), col n = lane%16
        float* sc = lds + key0 + lane_m;
#pragma unroll
        for (int r = 0; r < 8; ++r)
            sc[(size_t)(r + 8 * lane_h) * ROWSTRIDE] = c[r] * scale;
    }
    __syncthreads();

    // ---- Softmax over each row of 2048 (16 threads per row) ---------------
    {
        const int row  = tid >> 4;               // 0..15
        const int colg = tid & 15;               // 0..15
        float* srow = lds + (size_t)row * ROWSTRIDE;

        float mx = -INFINITY;
        for (int j = colg; j < SEQ; j += 16) mx = fmaxf(mx, srow[j]);
#pragma unroll
        for (int off = 8; off >= 1; off >>= 1)
            mx = fmaxf(mx, __shfl_xor(mx, off, 16));

        float sum = 0.0f;
        for (int j = colg; j < SEQ; j += 16) {
            float e = __expf(srow[j] - mx);
            srow[j] = e;
            sum += e;
        }
#pragma unroll
        for (int off = 8; off >= 1; off >>= 1)
            sum += __shfl_xor(sum, off, 16);
        const float inv = 1.0f / sum;

        float* arow = attn + ((size_t)b * SEQ + (size_t)qt * QTILE + row) * SEQ;
        for (int j = colg; j < SEQ; j += 16) {
            float p = srow[j] * inv;
            srow[j] = p;                          // keep normalized probs in LDS
            arow[j] = p;                          // coalesced attn output
        }
    }
    __syncthreads();

    // ---- Pass 2: context = P V  (each wave: one 16x16 output tile) --------
    const int col0 = wave * 16;                  // SGPR
    const float* prow  = lds + (size_t)lane_m * ROWSTRIDE + 16 * lane_h;
    const float* vbase = Vb + (size_t)(16 * lane_h) * DIM + col0 + lane_m;
    v8f acc = {};
#pragma unroll 2
    for (int kt = 0; kt < SEQ / 32; ++kt) {
        const int k0 = kt * 32;
        // A fragment: probabilities P[m, k0 + ...] gathered from LDS (f32->f16)
        const float4* p0 = reinterpret_cast<const float4*>(prow + k0);
        const float4* p1 = reinterpret_cast<const float4*>(prow + k0 + 16);
        v16h ap = pack_h16(p0[0], p0[1], p0[2], p0[3]);
        // B fragment: V[k0 + 16*h + e, col0 + lane_m], stride-DIM dword reads
        const float* vc = vbase + (size_t)k0 * DIM;
        v16h bv;
#pragma unroll
        for (int e = 0; e < 16; ++e)
            bv[e] = (_Float16)vc[(size_t)e * DIM];
        acc = __builtin_amdgcn_wmma_f32_16x16x32_f16(
                  false, ap, false, bv, (short)0, acc, false, false);
    }
    float* crow = ctx + ((size_t)b * SEQ + (size_t)qt * QTILE) * DIM;
#pragma unroll
    for (int r = 0; r < 8; ++r)
        crow[(size_t)(r + 8 * lane_h) * DIM + col0 + lane_m] = acc[r];
}

extern "C" void kernel_launch(void* const* d_in, const int* in_sizes, int n_in,
                              void* d_out, int out_size, void* d_ws, size_t ws_size,
                              hipStream_t stream) {
    (void)in_sizes; (void)n_in; (void)out_size; (void)d_ws; (void)ws_size;
    const float* Q = (const float*)d_in[0];
    const float* K = (const float*)d_in[1];
    const float* V = (const float*)d_in[2];
    float* ctx  = (float*)d_out;                                   // [B,S,D]
    float* attn = (float*)d_out + (size_t)BATCH * SEQ * DIM;       // [B,S,S]

    // allow >64KB dynamic LDS (320KB/WGP on CDNA5); harmless if already set
    (void)hipFuncSetAttribute((const void*)sdpa_wmma_kernel,
                              hipFuncAttributeMaxDynamicSharedMemorySize,
                              LDS_BYTES);

    dim3 grid(BATCH * (SEQ / QTILE));            // 2048 workgroups
    dim3 block(THREADS);                         // 8 wave32
    sdpa_wmma_kernel<<<grid, block, LDS_BYTES, stream>>>(Q, K, V, ctx, attn);
}
